// EntityAwareBertSelfAttention_13030930776490
// MI455X (gfx1250) — matrix operand order repaired and checked by reading, not compiled, not run on hardware
//
#include <hip/hip_runtime.h>
#include <hip/hip_bf16.h>

typedef __attribute__((ext_vector_type(16))) _Float16 v16h;
typedef __attribute__((ext_vector_type(8)))  _Float16 v8h;
typedef __attribute__((ext_vector_type(8)))  float    v8f;

static constexpr int BB = 4;
static constexpr int TT = 2048;
static constexpr int EE = 512;
static constexpr int DD = 768;
static constexpr int HH = 12;
static constexpr int DH = 64;
static constexpr int SS = TT + EE;   // 2560

// ---------------------------------------------------------------------------
// WMMA helpers (gfx1250, wave32). D = A(16x32) * B(32x16) + C(16x16 f32)
// A layout (16-bit): lanes 0-15 hold row M=lane, halves = K {0..7, 16..23};
//                    lanes 16-31 hold row M=lane-16, halves = K {8..15, 24..31}.
// B layout (16-bit): lanes 0-15 hold col N=lane, halves = K {0..7, 16..23};
//                    lanes 16-31 hold col N=lane-16, halves = K {8..15, 24..31}.
// C/D layout (f32):  VGPR i: lanes 0-15 -> (M=i, N=lane); lanes 16-31 -> (M=i+8).
// ---------------------------------------------------------------------------

__device__ __forceinline__ v8f wmma_f16(v16h a, v16h b, v8f c) {
  return __builtin_amdgcn_wmma_f32_16x16x32_f16(false, a, false, b,
                                                (short)0, c, false, false);
}

// ---- DPP16 butterfly reductions across 16 lanes (halves independent) ------
// Single-instruction fused DPP arithmetic: dst = op(dpp(src0), src1).
#define RED_STEP_MAX(CTRL)                                                    \
  { float y_;                                                                 \
    asm("v_max_num_f32_dpp %0, %1, %2 " CTRL                                  \
        " row_mask:0xf bank_mask:0xf bound_ctrl:1"                            \
        : "=v"(y_) : "v"(x), "v"(x));                                         \
    x = y_; }
#define RED_STEP_ADD(CTRL)                                                    \
  { float y_;                                                                 \
    asm("v_add_f32_dpp %0, %1, %2 " CTRL                                      \
        " row_mask:0xf bank_mask:0xf bound_ctrl:1"                            \
        : "=v"(y_) : "v"(x), "v"(x));                                         \
    x = y_; }

__device__ __forceinline__ float red16_max(float x) {
  RED_STEP_MAX("quad_perm:[1,0,3,2]")   // xor 1
  RED_STEP_MAX("quad_perm:[2,3,0,1]")   // xor 2
  RED_STEP_MAX("row_half_mirror")       // xor 4 (quad-uniform by now)
  RED_STEP_MAX("row_mirror")            // xor 8 (oct-uniform by now)
  return x;
}
__device__ __forceinline__ float red16_sum(float x) {
  RED_STEP_ADD("quad_perm:[1,0,3,2]")
  RED_STEP_ADD("quad_perm:[2,3,0,1]")
  RED_STEP_ADD("row_half_mirror")
  RED_STEP_ADD("row_mirror")
  return x;
}

// Load A fragment from a row-major f16 matrix: rows row0..row0+15, cols k0..k0+31.
__device__ __forceinline__ v16h load_a(const _Float16* __restrict__ base,
                                       int row0, int lda, int k0, int lane) {
  const int l  = lane & 15;
  const int hv = lane >> 4;
  const _Float16* p = base + (size_t)(row0 + l) * lda + k0 + hv * 8;
  v8h lo = *(const v8h*)p;          // K offsets {0..7} or {8..15}
  v8h hi = *(const v8h*)(p + 16);   // K offsets {16..23} or {24..31}
  v16h r;
#pragma unroll
  for (int i = 0; i < 8; ++i) { r[i] = lo[i]; r[i + 8] = hi[i]; }
  return r;
}

// Load B fragment: B[k][n] = M[row0+n][k0+k] for a row-major f16 matrix M
// (i.e. column n of B is row row0+n of M). K {0..7,16..23} or {8..15,24..31} per lane.
__device__ __forceinline__ v16h load_b(const _Float16* __restrict__ base,
                                       int row0, int lda, int k0, int lane) {
  const int l  = lane & 15;
  const int hv = lane >> 4;
  const _Float16* p = base + (size_t)(row0 + l) * lda + k0 + hv * 8;
  v8h lo = *(const v8h*)p;
  v8h hi = *(const v8h*)(p + 16);
  v16h r;
#pragma unroll
  for (int i = 0; i < 8; ++i) { r[i] = lo[i]; r[i + 8] = hi[i]; }
  return r;
}

// ---------------------------------------------------------------------------
// Prep kernels
// ---------------------------------------------------------------------------
__global__ void k_convert(const float* __restrict__ src,
                          _Float16* __restrict__ dst, int n) {
  int i = blockIdx.x * blockDim.x + threadIdx.x;
  if (i < n) dst[i] = (_Float16)src[i];
}

__global__ void k_pae(const float* __restrict__ ent, const float* __restrict__ qpos,
                      _Float16* __restrict__ dst, int n) {
  int i = blockIdx.x * blockDim.x + threadIdx.x;
  if (i < n) dst[i] = (_Float16)((ent[i] + qpos[i]) * 0.5f);
}

// ---------------------------------------------------------------------------
// Projection: Y[m,n] = sum_k X[m,k] * W[n,k] + bias[n].
// One wave computes a 16x64 tile == all of one head's DH for 16 rows:
// 4 WMMAs per k-step, A fragment reused 4x, 96 WMMAs per wave.
// LROWS is a compile-time power of two -> div/mod become shift/and.
// VMODE 0: write f16 into [B,H,S,DH] at s = s_off + (m % LROWS)
// VMODE 1: write f16 into [B,H,DH,S] (transposed V)
// ---------------------------------------------------------------------------
template <int LROWS, int VMODE>
__global__ void k_proj(const _Float16* __restrict__ X, const _Float16* __restrict__ W,
                       const float* __restrict__ bias, _Float16* __restrict__ out,
                       int s_off) {
  const int lane = threadIdx.x;
  const int tile = blockIdx.x;
  const int m0 = (tile / HH) * 16;   // 12 head-tiles per row-tile
  const int h  = tile % HH;          // output head; n0 = h*64

  v8f c[4] = {v8f{}, v8f{}, v8f{}, v8f{}};
  for (int kk = 0; kk < DD; kk += 32) {
    const v16h a = load_a(X, m0, DD, kk, lane);
#pragma unroll
    for (int j = 0; j < 4; ++j) {
      const v16h b = load_b(W, h * 64 + j * 16, DD, kk, lane);
      c[j] = wmma_f16(a, b, c[j]);
    }
  }

  const int l  = lane & 15;
  const int hv = lane >> 4;
#pragma unroll
  for (int j = 0; j < 4; ++j) {
    const int dh = j * 16 + l;
    const float bv = bias[h * 64 + dh];
#pragma unroll
    for (int i = 0; i < 8; ++i) {
      const int m    = m0 + i + 8 * hv;
      const int bidx = m / LROWS;          // shift (LROWS = 2048 or 512)
      const int s    = s_off + (m & (LROWS - 1));
      const float val = c[j][i] + bv;
      size_t idx;
      if (VMODE) idx = ((size_t)(bidx * HH + h) * DH + dh) * SS + s;
      else       idx = ((size_t)(bidx * HH + h) * SS + s) * DH + dh;
      out[idx] = (_Float16)val;
    }
  }
}

// ---------------------------------------------------------------------------
// Flash attention: one wave handles 16 queries for one (b,h), streaming the
// full S=2560 keys in 32-key chunks with online softmax.
// ---------------------------------------------------------------------------
__global__ void k_attn(const _Float16* __restrict__ Q, const _Float16* __restrict__ K,
                       const _Float16* __restrict__ Vt, const float* __restrict__ mask,
                       float* __restrict__ out) {
  __shared__ __align__(16) _Float16 P[16 * 32];

  const int lane = threadIdx.x;
  const int l    = lane & 15;
  const int hv   = lane >> 4;

  const int qtiles = SS / 16;          // 160
  const int tile = blockIdx.x;
  const int qt = tile % qtiles;
  const int bh = tile / qtiles;
  const int h  = bh % HH;
  const int b  = bh / HH;
  const int s0 = qt * 16;

  const _Float16* Qb = Q  + (size_t)bh * SS * DH;
  const _Float16* Kb = K  + (size_t)bh * SS * DH;
  const _Float16* Vb = Vt + (size_t)bh * DH * SS;
  const float*    mk = mask + (size_t)b * SS;

  const v16h qf0 = load_a(Qb, s0, DH, 0,  lane);
  const v16h qf1 = load_a(Qb, s0, DH, 32, lane);

  v8f acc0 = {}, acc1 = {}, acc2 = {}, acc3 = {};
  float m_i[8], l_i[8];
#pragma unroll
  for (int i = 0; i < 8; ++i) { m_i[i] = -3.0e38f; l_i[i] = 0.0f; }
  const float scale = 0.125f;          // 1/sqrt(64)

  for (int k0 = 0; k0 < SS; k0 += 32) {
    // prefetch two chunks ahead (global_prefetch_b8); one line per lane
    if (k0 + 64 < SS) {
      __builtin_prefetch(Kb + (size_t)(k0 + 64 + lane) * DH, 0, 3);
      __builtin_prefetch(Vb + (size_t)lane * SS + k0 + 64, 0, 3);
      __builtin_prefetch(Vb + (size_t)(lane + 32) * SS + k0 + 64, 0, 3);
    }

    // ---- scores: two 16x16 key tiles, inner dim DH=64 (2 WMMAs each) ----
    v8f sc0 = {}, sc1 = {};
    sc0 = wmma_f16(qf0, load_b(Kb, k0,      DH, 0,  lane), sc0);
    sc0 = wmma_f16(qf1, load_b(Kb, k0,      DH, 32, lane), sc0);
    sc1 = wmma_f16(qf0, load_b(Kb, k0 + 16, DH, 0,  lane), sc1);
    sc1 = wmma_f16(qf1, load_b(Kb, k0 + 16, DH, 32, lane), sc1);

    const float am0 = mk[k0 + l];
    const float am1 = mk[k0 + 16 + l];

    __syncthreads();   // previous iteration's P fragment fully consumed
#pragma unroll
    for (int i = 0; i < 8; ++i) {
      float x0 = sc0[i] * scale + am0;
      float x1 = sc1[i] * scale + am1;
      const float mx = red16_max(fmaxf(x0, x1));          // fused DPP max
      const float mn    = fmaxf(m_i[i], mx);
      const float alpha = __expf(m_i[i] - mn);
      const float e0 = __expf(x0 - mn);
      const float e1 = __expf(x1 - mn);
      const float rs = red16_sum(e0 + e1);                // fused DPP add
      l_i[i] = l_i[i] * alpha + rs;
      m_i[i] = mn;
      acc0[i] *= alpha; acc1[i] *= alpha; acc2[i] *= alpha; acc3[i] *= alpha;
      // stage probs row-major [16][32] for the A-fragment reload
      const int r = i + 8 * hv;
      P[r * 32 + l]      = (_Float16)e0;
      P[r * 32 + 16 + l] = (_Float16)e1;
    }
    __syncthreads();

    // ---- P(16x32) * V(32x64): 4 d-tiles ----
    const v16h pf = load_a(&P[0], 0, 32, 0, lane);
    acc0 = wmma_f16(pf, load_b(Vb, 0,  SS, k0, lane), acc0);
    acc1 = wmma_f16(pf, load_b(Vb, 16, SS, k0, lane), acc1);
    acc2 = wmma_f16(pf, load_b(Vb, 32, SS, k0, lane), acc2);
    acc3 = wmma_f16(pf, load_b(Vb, 48, SS, k0, lane), acc3);
  }

  // ---- finalize: divide by row sums, scatter to [B,S,D] split output ----
#pragma unroll
  for (int i = 0; i < 8; ++i) {
    const int r = i + 8 * hv;
    const int s = s0 + r;
    const float inv = 1.0f / l_i[i];
    size_t base;
    if (s < TT) base = ((size_t)b * TT + s) * DD + h * DH;
    else        base = (size_t)BB * TT * DD + ((size_t)b * EE + (s - TT)) * DD + h * DH;
    out[base + 0  + l] = acc0[i] * inv;
    out[base + 16 + l] = acc1[i] * inv;
    out[base + 32 + l] = acc2[i] * inv;
    out[base + 48 + l] = acc3[i] * inv;
  }
}

// ---------------------------------------------------------------------------
// Launch
// ---------------------------------------------------------------------------
extern "C" void kernel_launch(void* const* d_in, const int* in_sizes, int n_in,
                              void* d_out, int out_size, void* d_ws, size_t ws_size,
                              hipStream_t stream) {
  (void)in_sizes; (void)n_in; (void)out_size; (void)ws_size;

  const float* tok  = (const float*)d_in[0];
  const float* ent  = (const float*)d_in[1];
  const float* mask = (const float*)d_in[2];
  const float* qpos = (const float*)d_in[3];
  const float* Wf[6] = { (const float*)d_in[4],  (const float*)d_in[6],
                         (const float*)d_in[8],  (const float*)d_in[10],
                         (const float*)d_in[12], (const float*)d_in[14] }; // Wq Wk Wv Weq Wek Wev
  const float* bf[6] = { (const float*)d_in[5],  (const float*)d_in[7],
                         (const float*)d_in[9],  (const float*)d_in[11],
                         (const float*)d_in[13], (const float*)d_in[15] };

  _Float16* ws = (_Float16*)d_ws;
  size_t o = 0;
  _Float16* tok16 = ws + o; o += (size_t)BB * TT * DD;
  _Float16* pae16 = ws + o; o += (size_t)BB * EE * DD;
  _Float16* ent16 = ws + o; o += (size_t)BB * EE * DD;
  _Float16* W16[6];
  for (int i = 0; i < 6; ++i) { W16[i] = ws + o; o += (size_t)DD * DD; }
  _Float16* Qb  = ws + o; o += (size_t)BB * HH * SS * DH;
  _Float16* Kb  = ws + o; o += (size_t)BB * HH * SS * DH;
  _Float16* Vtb = ws + o; o += (size_t)BB * HH * SS * DH;

  const int nTok = BB * TT * DD;
  const int nEnt = BB * EE * DD;
  const int nW   = DD * DD;

  k_convert<<<(nTok + 255) / 256, 256, 0, stream>>>(tok, tok16, nTok);
  k_convert<<<(nEnt + 255) / 256, 256, 0, stream>>>(ent, ent16, nEnt);
  k_pae    <<<(nEnt + 255) / 256, 256, 0, stream>>>(ent, qpos, pae16, nEnt);
  for (int i = 0; i < 6; ++i)
    k_convert<<<(nW + 255) / 256, 256, 0, stream>>>(Wf[i], W16[i], nW);

  const dim3 wv(32);
  const int gT = (BB * TT / 16) * HH;   // 16x64 tiles: (M/16) * (768/64)
  const int gE = (BB * EE / 16) * HH;
  // Q
  k_proj<TT, 0><<<gT, wv, 0, stream>>>(tok16, W16[0], bf[0], Qb,  0);
  k_proj<EE, 0><<<gE, wv, 0, stream>>>(pae16, W16[3], bf[3], Qb,  TT);
  // K
  k_proj<TT, 0><<<gT, wv, 0, stream>>>(tok16, W16[1], bf[1], Kb,  0);
  k_proj<EE, 0><<<gE, wv, 0, stream>>>(pae16, W16[4], bf[4], Kb,  TT);
  // V (transposed layout; entities use raw entity states, not pae)
  k_proj<TT, 1><<<gT, wv, 0, stream>>>(tok16, W16[2], bf[2], Vtb, 0);
  k_proj<EE, 1><<<gE, wv, 0, stream>>>(ent16, W16[5], bf[5], Vtb, TT);

  k_attn<<<BB * HH * (SS / 16), wv, 0, stream>>>(Qb, Kb, Vtb, mask, (float*)d_out);
}
